// AttentionActPrune_14611478741379
// MI455X (gfx1250) — compile-verified, matrix operand order checked by reading, not run on hardware
//
#include <hip/hip_runtime.h>
#include <hip/hip_bf16.h>

// ---------------------------------------------------------------------------
// ViT attention (B=64, S=577, D=768, H=12, Dh=64) on gfx1250 (CDNA5, wave32).
// All GEMMs use v_wmma_f32_16x16x32_f16 (f16 inputs, f32 accumulate).
// Pipeline:
//   1) cvt f32->f16 of hidden + Wq/Wk/Wv/Wo into workspace
//   2) WMMA GEMM (128x128 block tile, 32x64 wave tile, 8 WMMA/k-step):
//        Q/K/V = Xh @ W.T + b  -> f16 [B,S,768]
//   3) flash attention (online softmax, scores never leave registers/LDS)
//        -> ctx f16 [B,S,768]
//   4) WMMA GEMM: out = ctx @ Wo.T + bo -> f32 [B,S,768] (d_out)
// Workspace: 5 * 56.7MB (Xh,Q,K,V,ctx) + 4 * 1.18MB (weights) ~ 289 MB.
// ---------------------------------------------------------------------------

typedef _Float16 f16;
typedef __attribute__((ext_vector_type(8)))  _Float16 v8h;
typedef __attribute__((ext_vector_type(16))) _Float16 v16h;
typedef __attribute__((ext_vector_type(8)))  float    v8f;

#define NB   64
#define NS   577
#define ND   768
#define NH   12
#define NDH  64
#define NM   36928   // NB*NS

static __device__ __forceinline__ v8h ld8(const f16* p) { return *(const v8h*)p; }

static __device__ __forceinline__ v16h cat16(v8h a, v8h b) {
  return __builtin_shufflevector(a, b, 0,1,2,3,4,5,6,7,8,9,10,11,12,13,14,15);
}

static __device__ __forceinline__ v8f wmma_f16(v16h a, v16h b, v8f c) {
  // D = A(16x32 f16) x B(32x16 f16) + C(16x16 f32)
  return __builtin_amdgcn_wmma_f32_16x16x32_f16(false, a, false, b, (short)0, c,
                                                false, false);
}

static __device__ __forceinline__ v8f splat8(float x) {
  v8f r;
#pragma unroll
  for (int i = 0; i < 8; ++i) r[i] = x;
  return r;
}

// ---------------------------------------------------------------------------
// f32 -> f16 conversion (vectorized x4; all sizes are multiples of 4)
// ---------------------------------------------------------------------------
__global__ void cvt_f32_to_f16(const float* __restrict__ x, f16* __restrict__ y,
                               int n) {
  int i = (blockIdx.x * blockDim.x + threadIdx.x) * 4;
  const int stride = gridDim.x * blockDim.x * 4;
  for (; i + 3 < n; i += stride) {
    float4 v = *(const float4*)(x + i);
    y[i + 0] = (f16)v.x;
    y[i + 1] = (f16)v.y;
    y[i + 2] = (f16)v.z;
    y[i + 3] = (f16)v.w;
  }
}

// ---------------------------------------------------------------------------
// WMMA GEMM: C[M,768] = X[M,768] @ W[768,768]^T + bias
//   block = 256 threads (8 waves), tile 128(M) x 128(N), K-step 32.
//   wave (wm = wid&3 -> 32 rows, wn = wid>>2 -> 64 cols) computes 32x64:
//     2 A-frags + 4 B-frags -> 8 WMMAs per k-step (1.5 ds_load_b128 / WMMA).
//   MODE 0: store f16 [M,768];  MODE 1: store f32 [M,768]  (compile-time).
// N=768 is a multiple of 128; M=36928 is not -> clamp loads; only the last
// M-tile takes the predicated store path (uniform branch).
// ---------------------------------------------------------------------------
template <int MODE>
__global__ __launch_bounds__(256) void gemm_wmma(const f16* __restrict__ X,
                                                 const f16* __restrict__ W,
                                                 const float* __restrict__ bias,
                                                 void* __restrict__ out) {
  __shared__ f16 As[128 * 40];  // 128 rows x 32 halves, stride 40 (16B aligned)
  __shared__ f16 Bs[128 * 40];

  const int m0   = blockIdx.x * 128;
  const int n0   = blockIdx.y * 128;
  const int tid  = threadIdx.x;
  const int wid  = tid >> 5;
  const int lane = tid & 31;
  const int l16  = lane & 15;
  const int hi   = lane >> 4;       // which K-half group this lane owns
  const int wm   = wid & 3;         // M strip 0..3 (32 rows each)
  const int wn   = wid >> 2;        // N strip 0..1 (64 cols each)

  const int lrow = tid >> 1;        // 0..127: LDS fill row
  const int lseg = (tid & 1) * 16;  // 0 or 16: 16-half segment

  int am = m0 + lrow; if (am > NM - 1) am = NM - 1;   // clamp OOB M rows
  const f16* xrow = X + (size_t)am * ND;
  const f16* wrow = W + (size_t)(n0 + lrow) * ND;

  v8f acc[2][4];
#pragma unroll
  for (int i = 0; i < 2; ++i)
#pragma unroll
    for (int j = 0; j < 4; ++j) acc[i][j] = splat8(0.f);

  for (int kb = 0; kb < ND; kb += 32) {
    __syncthreads();
    *(v8h*)&As[lrow * 40 + lseg]     = ld8(xrow + kb + lseg);
    *(v8h*)&As[lrow * 40 + lseg + 8] = ld8(xrow + kb + lseg + 8);
    *(v8h*)&Bs[lrow * 40 + lseg]     = ld8(wrow + kb + lseg);
    *(v8h*)&Bs[lrow * 40 + lseg + 8] = ld8(wrow + kb + lseg + 8);
    __syncthreads();

    // A fragments: lane = row (M); halves {0-7,16-23} (lanes<16) / {8-15,24-31}
    v16h a[2];
#pragma unroll
    for (int i = 0; i < 2; ++i) {
      const f16* ap = &As[(wm * 32 + i * 16 + l16) * 40 + hi * 8];
      a[i] = cat16(ld8(ap), ld8(ap + 16));
    }
    // B fragments: lane = col (N = W row), same K packing
    v16h bf[4];
#pragma unroll
    for (int j = 0; j < 4; ++j) {
      const f16* bp = &Bs[(wn * 64 + j * 16 + l16) * 40 + hi * 8];
      bf[j] = cat16(ld8(bp), ld8(bp + 16));
    }
#pragma unroll
    for (int i = 0; i < 2; ++i)
#pragma unroll
      for (int j = 0; j < 4; ++j) acc[i][j] = wmma_f16(a[i], bf[j], acc[i][j]);
  }

  // Epilogue. D layout: lane<16 -> row = base+r, lane>=16 -> row = base+8+r;
  // column = lane%16 within each 16-wide tile. Fast path (no predication)
  // unless this is the last M-tile (uniform branch).
  if (m0 + 128 <= NM) {
#pragma unroll
    for (int i = 0; i < 2; ++i) {
      const int mbase = m0 + wm * 32 + i * 16 + hi * 8;
#pragma unroll
      for (int j = 0; j < 4; ++j) {
        const int n    = n0 + wn * 64 + j * 16 + l16;
        const float bn = bias[n];
#pragma unroll
        for (int r = 0; r < 8; ++r) {
          const float v = acc[i][j][r] + bn;
          if (MODE == 0)
            ((f16*)out)[(size_t)(mbase + r) * ND + n] = (f16)v;
          else
            ((float*)out)[(size_t)(mbase + r) * ND + n] = v;
        }
      }
    }
  } else {
#pragma unroll
    for (int i = 0; i < 2; ++i) {
      const int mbase = m0 + wm * 32 + i * 16 + hi * 8;
#pragma unroll
      for (int j = 0; j < 4; ++j) {
        const int n    = n0 + wn * 64 + j * 16 + l16;
        const float bn = bias[n];
#pragma unroll
        for (int r = 0; r < 8; ++r) {
          if (mbase + r < NM) {
            const float v = acc[i][j][r] + bn;
            if (MODE == 0)
              ((f16*)out)[(size_t)(mbase + r) * ND + n] = (f16)v;
            else
              ((float*)out)[(size_t)(mbase + r) * ND + n] = v;
          }
        }
      }
    }
  }
}

// ---------------------------------------------------------------------------
// Flash attention: grid (ceil(S/128)=5, H=12, B=64), 256 threads = 8 waves.
// Q/K/V are f16 [B,S,768]; head h occupies columns [h*64, h*64+64).
// Each wave owns 16 query rows; key blocks of 64 staged in LDS
// (K row-major for QK^T B-frags, V transposed so PV B-frags are
// key-contiguous). softmax scale (1/8, a power of two -> exact in f16) is
// folded into the Q fragments once. Online softmax in the WMMA C layout;
// P converted D-layout -> A-layout through per-wave LDS scratch.
// ---------------------------------------------------------------------------
__global__ __launch_bounds__(256) void attn_flash(const f16* __restrict__ Q,
                                                  const f16* __restrict__ K,
                                                  const f16* __restrict__ V,
                                                  f16* __restrict__ ctx_out) {
  __shared__ f16 Kt[64 * 72];       // [key][dh]   stride 72 halves (144B)
  __shared__ f16 Vt[64 * 72];       // [dh][key]   (transposed)
  __shared__ f16 Pl[8 * 16 * 72];   // per-wave 16x64 P scratch

  const int b    = blockIdx.z;
  const int h    = blockIdx.y;
  const int tid  = threadIdx.x;
  const int wid  = tid >> 5;
  const int lane = tid & 31;
  const int l16  = lane & 15;
  const int hi   = lane >> 4;
  const int q0   = blockIdx.x * 128 + wid * 16;

  const f16* Qb = Q + (size_t)b * NS * ND + h * NDH;   // row stride ND
  const f16* Kb = K + (size_t)b * NS * ND + h * NDH;
  const f16* Vb = V + (size_t)b * NS * ND + h * NDH;

  // Q A-fragments (16 rows x 64 K), scale 1/sqrt(64)=0.125 folded in (exact).
  int qr = q0 + l16; if (qr > NS - 1) qr = NS - 1;
  const f16* qp = Qb + (size_t)qr * ND + hi * 8;
  const f16 sc = (f16)0.125f;
  v16h qa0 = cat16(ld8(qp),      ld8(qp + 16)) * sc;   // k = 0..31
  v16h qa1 = cat16(ld8(qp + 32), ld8(qp + 48)) * sc;   // k = 32..63

  v8f acc[4];
#pragma unroll
  for (int j = 0; j < 4; ++j) acc[j] = splat8(0.f);
  v8f mrow = splat8(-1e30f);
  v8f lsum = splat8(0.f);

  const int ldrow = tid >> 2;         // 0..63 key row to stage
  const int ldseg = (tid & 3) * 16;   // 16-half segment of dh

  for (int kb = 0; kb < NS; kb += 64) {
    __syncthreads();
    int krow = kb + ldrow; if (krow > NS - 1) krow = NS - 1;
    const f16* kp = Kb + (size_t)krow * ND + ldseg;
    const f16* vp = Vb + (size_t)krow * ND + ldseg;
    v8h k0 = ld8(kp), k1 = ld8(kp + 8);
    v8h v0 = ld8(vp), v1 = ld8(vp + 8);
    *(v8h*)&Kt[ldrow * 72 + ldseg]     = k0;
    *(v8h*)&Kt[ldrow * 72 + ldseg + 8] = k1;
#pragma unroll
    for (int j = 0; j < 8; ++j) {                 // transpose V into LDS
      Vt[(ldseg + j) * 72 + ldrow]     = v0[j];
      Vt[(ldseg + 8 + j) * 72 + ldrow] = v1[j];
    }
    __syncthreads();

    // ---- scores S = (Q/8) K^T : 4 tiles of 16 keys -----------------------
    v8f s[4];
#pragma unroll
    for (int t = 0; t < 4; ++t) {
      const f16* krp = &Kt[(t * 16 + l16) * 72 + hi * 8];
      v16h b0 = cat16(ld8(krp),      ld8(krp + 16));
      v16h b1 = cat16(ld8(krp + 32), ld8(krp + 48));
      v8f c = splat8(0.f);
      c = wmma_f16(qa0, b0, c);
      c = wmma_f16(qa1, b1, c);
      const int gk = kb + t * 16 + l16;           // key id is per-lane uniform
      s[t] = (gk > NS - 1) ? splat8(-1e30f) : c;
    }

    // ---- row max (rows live in VGPR index; reduce across each 16-lane half)
    v8f mx = s[0];
#pragma unroll
    for (int t = 1; t < 4; ++t)
#pragma unroll
      for (int r = 0; r < 8; ++r) mx[r] = fmaxf(mx[r], s[t][r]);
#pragma unroll
    for (int d = 1; d < 16; d <<= 1)
#pragma unroll
      for (int r = 0; r < 8; ++r) mx[r] = fmaxf(mx[r], __shfl_xor(mx[r], d, 32));

    v8f mnew, corr;
#pragma unroll
    for (int r = 0; r < 8; ++r) {
      mnew[r] = fmaxf(mrow[r], mx[r]);
      corr[r] = __expf(mrow[r] - mnew[r]);
    }
    mrow = mnew;

    v8f rs = splat8(0.f);
#pragma unroll
    for (int t = 0; t < 4; ++t)
#pragma unroll
      for (int r = 0; r < 8; ++r) {
        s[t][r] = __expf(s[t][r] - mnew[r]);
        rs[r]  += s[t][r];
      }
#pragma unroll
    for (int d = 1; d < 16; d <<= 1)
#pragma unroll
      for (int r = 0; r < 8; ++r) rs[r] += __shfl_xor(rs[r], d, 32);

#pragma unroll
    for (int r = 0; r < 8; ++r) lsum[r] = lsum[r] * corr[r] + rs[r];
#pragma unroll
    for (int j = 0; j < 4; ++j) acc[j] *= corr;

    // ---- P: D-layout -> A-layout via wave-private LDS scratch -------------
    f16* pw = &Pl[wid * 16 * 72];
#pragma unroll
    for (int t = 0; t < 4; ++t)
#pragma unroll
      for (int r = 0; r < 8; ++r)
        pw[(hi * 8 + r) * 72 + t * 16 + l16] = (f16)s[t][r];

    const f16* pr = pw + l16 * 72 + hi * 8;
    v16h pa0 = cat16(ld8(pr),      ld8(pr + 16));   // keys 0..31
    v16h pa1 = cat16(ld8(pr + 32), ld8(pr + 48));   // keys 32..63

    // ---- ctx += P V (B fragment: lane = dh, keys contiguous in Vt row) ----
#pragma unroll
    for (int j = 0; j < 4; ++j) {
      const f16* vrp = &Vt[(j * 16 + l16) * 72 + hi * 8];
      v16h vb0 = cat16(ld8(vrp),      ld8(vrp + 16));
      v16h vb1 = cat16(ld8(vrp + 32), ld8(vrp + 48));
      acc[j] = wmma_f16(pa0, vb0, acc[j]);
      acc[j] = wmma_f16(pa1, vb1, acc[j]);
    }
  }

  // ---- finalize: ctx /= l, store f16 to [B, S, 768] -----------------------
  // Fast path: all 16 query rows valid (uniform; only last q-tile predicates).
  const int col0 = h * NDH;
  f16* outp = ctx_out + (size_t)b * NS * ND + col0;
  if (q0 + 16 <= NS) {
#pragma unroll
    for (int j = 0; j < 4; ++j)
#pragma unroll
      for (int r = 0; r < 8; ++r)
        outp[(size_t)(q0 + hi * 8 + r) * ND + j * 16 + l16] =
            (f16)(acc[j][r] / lsum[r]);
  } else {
#pragma unroll
    for (int j = 0; j < 4; ++j)
#pragma unroll
      for (int r = 0; r < 8; ++r) {
        const int q = q0 + hi * 8 + r;
        if (q < NS)
          outp[(size_t)q * ND + j * 16 + l16] = (f16)(acc[j][r] / lsum[r]);
      }
  }
}

// ---------------------------------------------------------------------------
extern "C" void kernel_launch(void* const* d_in, const int* in_sizes, int n_in,
                              void* d_out, int out_size, void* d_ws,
                              size_t ws_size, hipStream_t stream) {
  const float* hidden = (const float*)d_in[0];
  const float* Wq = (const float*)d_in[1];
  const float* bq = (const float*)d_in[2];
  const float* Wk = (const float*)d_in[3];
  const float* bk = (const float*)d_in[4];
  const float* Wv = (const float*)d_in[5];
  const float* bv = (const float*)d_in[6];
  const float* Wo = (const float*)d_in[7];
  const float* bo = (const float*)d_in[8];

  char* ws = (char*)d_ws;
  const size_t szX = (size_t)NM * ND * sizeof(f16);       // 56.7 MB
  const size_t szW = (size_t)ND * ND * sizeof(f16);       // 1.18 MB
  size_t off = 0;
  f16* Xh  = (f16*)(ws + off); off += szX;
  f16* Wqh = (f16*)(ws + off); off += szW;
  f16* Wkh = (f16*)(ws + off); off += szW;
  f16* Wvh = (f16*)(ws + off); off += szW;
  f16* Woh = (f16*)(ws + off); off += szW;
  f16* Qh  = (f16*)(ws + off); off += szX;
  f16* Kh  = (f16*)(ws + off); off += szX;
  f16* Vh  = (f16*)(ws + off); off += szX;
  f16* Ch  = (f16*)(ws + off); off += szX;

  // 1) conversions to f16
  cvt_f32_to_f16<<<8192, 256, 0, stream>>>(hidden, Xh, NM * ND);
  cvt_f32_to_f16<<<576, 256, 0, stream>>>(Wq, Wqh, ND * ND);
  cvt_f32_to_f16<<<576, 256, 0, stream>>>(Wk, Wkh, ND * ND);
  cvt_f32_to_f16<<<576, 256, 0, stream>>>(Wv, Wvh, ND * ND);
  cvt_f32_to_f16<<<576, 256, 0, stream>>>(Wo, Woh, ND * ND);

  // 2) Q/K/V projections: tiles 128(M) x 128(N)
  dim3 gg((NM + 127) / 128, ND / 128);
  gemm_wmma<0><<<gg, 256, 0, stream>>>(Xh, Wqh, bq, Qh);
  gemm_wmma<0><<<gg, 256, 0, stream>>>(Xh, Wkh, bk, Kh);
  gemm_wmma<0><<<gg, 256, 0, stream>>>(Xh, Wvh, bv, Vh);

  // 3) flash attention -> ctx f16 [B,S,768]
  attn_flash<<<dim3((NS + 127) / 128, NH, NB), 256, 0, stream>>>(Qh, Kh, Vh, Ch);

  // 4) output projection -> f32 d_out
  gemm_wmma<1><<<gg, 256, 0, stream>>>(Ch, Woh, bo, d_out);
}
